// Interaction_layer_39934605918649
// MI455X (gfx1250) — matrix operand order repaired and checked by reading, not compiled, not run on hardware
//
#include <hip/hip_runtime.h>
#include <hip/hip_bf16.h>
#include <stdint.h>

// ---------- CDNA5 WMMA types ----------
typedef __attribute__((ext_vector_type(16))) __bf16   v16bf;
typedef __attribute__((ext_vector_type(8)))  float    v8f;
typedef __attribute__((ext_vector_type(4)))  uint32_t u32x4;
typedef __attribute__((ext_vector_type(4)))  float    f32x4;

union ABits { u32x4 q[2]; v16bf v; };   // 32B fragment built from two 16B LDS chunks
union BBits { uint32_t u[8]; v16bf v; };

__device__ __forceinline__ unsigned short f2bf(float x) {
    uint32_t u = __float_as_uint(x);
    uint32_t r = u + 0x7FFFu + ((u >> 16) & 1u);     // round-to-nearest-even
    return (unsigned short)(r >> 16);
}

// CDNA5 has a native V_TANH_F32 TRANS op — one instruction instead of the
// branchy libm expansion. Keep a fallback for older toolchains.
__device__ __forceinline__ float hw_tanh(float x) {
#if __has_builtin(__builtin_amdgcn_tanhf)
    return __builtin_amdgcn_tanhf(x);
#else
    return tanhf(x);
#endif
}
__device__ __forceinline__ float hw_sigmoid(float x) {
    return 0.5f * hw_tanh(0.5f * x) + 0.5f;
}

// Load WMMA A-fragment (16x32 bf16) for k-slice s from a row-major LDS tile.
// Lane<16: M=lane, K = 32s+[0..15]; Lane>=16: M=lane-16, K = 32s+[16..31].
template <int ROWLEN>
__device__ __forceinline__ v16bf loadAfrag(const unsigned short (*buf)[ROWLEN], int s, int lane) {
    ABits a;
    int row = lane & 15;
    int b1 = 32 * s + ((lane < 16) ? 0 : 16);
    int b2 = 32 * s + ((lane < 16) ? 8 : 24);
    a.q[0] = *(const u32x4*)&buf[row][b1];
    a.q[1] = *(const u32x4*)&buf[row][b2];
    return a.v;
}

// Pack a B-fragment lane: 16 consecutive K values (fp32 -> bf16) for one column n.
__device__ __forceinline__ v16bf packB(const float* __restrict__ src) {
    BBits b;
#pragma unroll
    for (int j = 0; j < 8; ++j) {
        uint32_t lo = f2bf(src[2 * j]);
        uint32_t hi = f2bf(src[2 * j + 1]);
        b.u[j] = lo | (hi << 16);
    }
    return b.v;
}
__device__ __forceinline__ v16bf packB2(const float* __restrict__ s0, const float* __restrict__ s1) {
    BBits b;
#pragma unroll
    for (int j = 0; j < 8; ++j) {
        uint32_t lo = f2bf(s0[2 * j] + s1[2 * j]);
        uint32_t hi = f2bf(s0[2 * j + 1] + s1[2 * j + 1]);
        b.u[j] = lo | (hi << 16);
    }
    return b.v;
}

#define WMMA_BF16(A, B, C) \
    __builtin_amdgcn_wmma_f32_16x16x32_bf16(false, (A), false, (B), (short)0, (C), false, false)

// ================= Kernel 1: conv1d(3->64,k=5,pad=2) + ReLU on x[:,0] =================
// act layout: (T=100, B=1024, 64) bf16, ready for batch-tiled row-major staging.
__global__ __launch_bounds__(256) void conv_kernel(const float* __restrict__ x,
                                                   const float* __restrict__ conv_w,
                                                   const float* __restrict__ conv_b,
                                                   unsigned short* __restrict__ act) {
    __shared__ float wsm[64 * 15];
    __shared__ float bsm[64];
    int tid = threadIdx.x;
    for (int i = tid; i < 960; i += 256) wsm[i] = conv_w[i];
    if (tid < 64) bsm[tid] = conv_b[tid];
    __syncthreads();

    int oc = tid & 63;
    int t  = blockIdx.y * 4 + (tid >> 6);          // 25*4 = 100 exactly
    int b  = blockIdx.x;
    const float* ag = x + (size_t)b * (128 * 3 * 100);   // x[b,0,:,:]

    float acc = bsm[oc];
#pragma unroll
    for (int c = 0; c < 3; ++c)
#pragma unroll
        for (int k = 0; k < 5; ++k) {
            int tt = t + k - 2;
            if (tt >= 0 && tt < 100) acc += ag[c * 100 + tt] * wsm[oc * 15 + c * 5 + k];
        }
    acc = fmaxf(acc, 0.0f);
    act[((size_t)t * 1024 + b) * 64 + oc] = f2bf(acc);
}

// ================= Kernel 2: fused LSTM scan, batch-tiled =================
// 64 workgroups x 512 threads (16 waves). Each WG owns 16 batch rows for all 100 steps.
// Per step: gates(16x512) = [xt|h](16x192) @ Wcombᵀ via bf16 WMMA, fp32 accum.
// Wave w -> N-tiles {2w, 2w+1}; B-fragments live in VGPRs for the whole kernel.
__global__ __launch_bounds__(512) void lstm_kernel(const unsigned short* __restrict__ act,
                                                   const float* __restrict__ w_ih,
                                                   const float* __restrict__ w_hh,
                                                   const float* __restrict__ b_ih,
                                                   const float* __restrict__ b_hh,
                                                   float* __restrict__ hout) {
    __shared__ alignas(16) unsigned short Abuf[16][192];  // [xt(0..63) | h(64..191)] bf16
    __shared__ alignas(16) float gbuf[16][512];           // gate pre-activations

    int tid  = threadIdx.x;
    int lane = tid & 31;
    int wave = tid >> 5;
    int b0   = blockIdx.x * 16;

    // --- Pin B fragments in VGPRs: 2 N-tiles x 6 k-slices (K = 64 ih + 128 hh) ---
    v16bf Bf[2][6];
#pragma unroll
    for (int ntl = 0; ntl < 2; ++ntl) {
        int n = (wave * 2 + ntl) * 16 + (lane & 15);
#pragma unroll
        for (int s = 0; s < 6; ++s) {
            int k0 = 32 * s + ((lane < 16) ? 0 : 16);
            const float* src = (k0 < 64) ? (w_ih + (size_t)n * 64 + k0)
                                         : (w_hh + (size_t)n * 128 + (k0 - 64));
            Bf[ntl][s] = packB(src);
        }
    }

    // --- Elementwise ownership: wave w owns row m=w; lane l owns hcols 4l..4l+3 ---
    int m_own = tid >> 5;
    int h0    = (lane * 4);
    f32x4 bias[4];
#pragma unroll
    for (int g = 0; g < 4; ++g) {
        f32x4 bi = *(const f32x4*)&b_ih[g * 128 + h0];
        f32x4 bh = *(const f32x4*)&b_hh[g * 128 + h0];
        bias[g] = bi + bh;
    }

    f32x4 c = {0.f, 0.f, 0.f, 0.f};

    // h_0 = 0
    for (int i = tid; i < 16 * 128; i += 512) Abuf[i >> 7][64 + (i & 127)] = 0;

    for (int t = 0; t < 100; ++t) {
        // Stage xt (16x64 bf16): one dword per thread.
        const uint32_t* srcu = (const uint32_t*)(act + ((size_t)t * 1024 + b0) * 64);
        {
            int m = tid >> 5, cp = tid & 31;
            ((uint32_t*)&Abuf[m][0])[cp] = srcu[m * 32 + cp];
        }
        if (t < 99 && lane == 0)
            __builtin_prefetch(act + ((size_t)(t + 1) * 1024 + b0) * 64, 0, 1);
        __syncthreads();   // xt staged + h_{t} from previous elementwise phase visible

        // --- WMMA: 2 N-tiles x 6 k-slices ---
        v8f acc0 = {0.f, 0.f, 0.f, 0.f, 0.f, 0.f, 0.f, 0.f};
        v8f acc1 = acc0;
#pragma unroll
        for (int s = 0; s < 6; ++s) {
            v16bf a = loadAfrag<192>(Abuf, s, lane);
            acc0 = WMMA_BF16(a, Bf[0][s], acc0);
            acc1 = WMMA_BF16(a, Bf[1][s], acc1);
        }
        {
            int mb = (lane < 16) ? 0 : 8;
            int nc0 = (wave * 2 + 0) * 16 + (lane & 15);
            int nc1 = (wave * 2 + 1) * 16 + (lane & 15);
#pragma unroll
            for (int r = 0; r < 8; ++r) {
                gbuf[mb + r][nc0] = acc0[r];
                gbuf[mb + r][nc1] = acc1[r];
            }
        }
        __syncthreads();   // gates complete (and all A-fragment reads done)

        // --- Gate math: c_t, h_t ; write h_t back to A-buffer as bf16 ---
        {
            f32x4 ig = *(const f32x4*)&gbuf[m_own][h0]       + bias[0];
            f32x4 fg = *(const f32x4*)&gbuf[m_own][128 + h0] + bias[1];
            f32x4 gg = *(const f32x4*)&gbuf[m_own][256 + h0] + bias[2];
            f32x4 og = *(const f32x4*)&gbuf[m_own][384 + h0] + bias[3];
            f32x4 hn;
#pragma unroll
            for (int j = 0; j < 4; ++j) {
                c[j] = hw_sigmoid(fg[j]) * c[j] + hw_sigmoid(ig[j]) * hw_tanh(gg[j]);
                hn[j] = hw_sigmoid(og[j]) * hw_tanh(c[j]);
                Abuf[m_own][64 + h0 + j] = f2bf(hn[j]);
            }
            if (t == 99)
                *(f32x4*)&hout[(size_t)(b0 + m_own) * 128 + h0] = hn;
        }
    }
}

// ================= Kernel 3: collapsed interaction + LN + linear + tile =================
// res = h@W0ᵀ + 127 * ( tanh(h @ (W1[:, :H]+W1[:, H:])ᵀ) @ W2ᵀ ) ; LN ; @lin_wᵀ+b ; tile 14x14.
__global__ __launch_bounds__(256) void inter_kernel(const float* __restrict__ hout,
                                                    const float* __restrict__ W1,
                                                    const float* __restrict__ W2,
                                                    const float* __restrict__ W0,
                                                    const float* __restrict__ ln_g,
                                                    const float* __restrict__ ln_b,
                                                    const float* __restrict__ lin_w,
                                                    const float* __restrict__ lin_b,
                                                    float* __restrict__ out) {
    __shared__ alignas(16) unsigned short Abuf[16][128];   // bf16 A operand
    __shared__ alignas(16) unsigned short Ibuf[16][128];   // tanh(inter) bf16
    __shared__ alignas(16) float resBuf[16][128];

    int tid = threadIdx.x, lane = tid & 31, wave = tid >> 5;  // 8 waves
    int b0 = blockIdx.x * 16;
    int n  = wave * 16 + (lane & 15);
    int mb = (lane < 16) ? 0 : 8;

    // stage h tile as bf16
    for (int i = tid; i < 16 * 128; i += 256)
        Abuf[i >> 7][i & 127] = f2bf(hout[(size_t)(b0 + (i >> 7)) * 128 + (i & 127)]);
    __syncthreads();

    v8f accW0 = {0.f, 0.f, 0.f, 0.f, 0.f, 0.f, 0.f, 0.f};
    v8f accW1 = accW0;
#pragma unroll
    for (int s = 0; s < 4; ++s) {
        int k0 = 32 * s + ((lane < 16) ? 0 : 16);
        v16bf a  = loadAfrag<128>(Abuf, s, lane);
        v16bf b0f = packB(W0 + (size_t)n * 128 + k0);
        v16bf b1f = packB2(W1 + (size_t)n * 256 + k0, W1 + (size_t)n * 256 + 128 + k0);
        accW0 = WMMA_BF16(a, b0f, accW0);
        accW1 = WMMA_BF16(a, b1f, accW1);
    }
    {   // tanh -> Ibuf (each wave writes its own 16 columns)
#pragma unroll
        for (int r = 0; r < 8; ++r) Ibuf[mb + r][n] = f2bf(hw_tanh(accW1[r]));
    }
    __syncthreads();

    v8f accW2 = {0.f, 0.f, 0.f, 0.f, 0.f, 0.f, 0.f, 0.f};
#pragma unroll
    for (int s = 0; s < 4; ++s) {
        int k0 = 32 * s + ((lane < 16) ? 0 : 16);
        v16bf a = loadAfrag<128>(Ibuf, s, lane);
        v16bf b = packB(W2 + (size_t)n * 128 + k0);
        accW2 = WMMA_BF16(a, b, accW2);
    }
    {   // res = h@W0ᵀ + 127 * inter
#pragma unroll
        for (int r = 0; r < 8; ++r) resBuf[mb + r][n] = accW0[r] + 127.0f * accW2[r];
    }
    __syncthreads();

    // --- LayerNorm: wave w handles rows 2w, 2w+1; wave32 shuffle reduction ---
#pragma unroll
    for (int rr = 0; rr < 2; ++rr) {
        int m = wave * 2 + rr;
        float v[4], s = 0.f, sq = 0.f;
#pragma unroll
        for (int j = 0; j < 4; ++j) { v[j] = resBuf[m][lane + 32 * j]; s += v[j]; sq += v[j] * v[j]; }
#pragma unroll
        for (int o = 16; o > 0; o >>= 1) { s += __shfl_xor(s, o, 32); sq += __shfl_xor(sq, o, 32); }
        float mu = s * (1.0f / 128.0f);
        float var = sq * (1.0f / 128.0f) - mu * mu;
        float rn = rsqrtf(var + 1e-5f);
#pragma unroll
        for (int j = 0; j < 4; ++j) {
            int col = lane + 32 * j;
            Abuf[m][col] = f2bf((v[j] - mu) * rn * ln_g[col] + ln_b[col]);
        }
    }
    __syncthreads();

    // --- final linear ---
    v8f accL = {0.f, 0.f, 0.f, 0.f, 0.f, 0.f, 0.f, 0.f};
#pragma unroll
    for (int s = 0; s < 4; ++s) {
        int k0 = 32 * s + ((lane < 16) ? 0 : 16);
        v16bf a = loadAfrag<128>(Abuf, s, lane);
        v16bf b = packB(lin_w + (size_t)n * 128 + k0);
        accL = WMMA_BF16(a, b, accL);
    }
    {
        float lb = lin_b[n];
#pragma unroll
        for (int r = 0; r < 8; ++r) resBuf[mb + r][n] = accL[r] + lb;
    }
    __syncthreads();

    // --- tiled store: out[b,14,14,128], 196 replicas per row, b128 stores ---
    for (int u = tid; u < 512; u += 256) {
        int m = u >> 5, q = u & 31;
        f32x4 val = *(const f32x4*)&resBuf[m][q * 4];
        float* dst = out + (size_t)(b0 + m) * 196 * 128 + q * 4;
        for (int ij = 0; ij < 196; ++ij) { *(f32x4*)dst = val; dst += 128; }
    }
}

// ================= Host launcher =================
extern "C" void kernel_launch(void* const* d_in, const int* in_sizes, int n_in,
                              void* d_out, int out_size, void* d_ws, size_t ws_size,
                              hipStream_t stream) {
    const float* x      = (const float*)d_in[0];
    const float* conv_w = (const float*)d_in[1];
    const float* conv_b = (const float*)d_in[2];
    const float* w_ih   = (const float*)d_in[3];
    const float* w_hh   = (const float*)d_in[4];
    const float* b_ih   = (const float*)d_in[5];
    const float* b_hh   = (const float*)d_in[6];
    const float* W1     = (const float*)d_in[7];
    const float* W2     = (const float*)d_in[8];
    const float* W0     = (const float*)d_in[9];
    const float* ln_g   = (const float*)d_in[10];
    const float* ln_b   = (const float*)d_in[11];
    const float* lin_w  = (const float*)d_in[12];
    const float* lin_b  = (const float*)d_in[13];
    float* out = (float*)d_out;

    // workspace: act (T,B,64) bf16 = 13,107,200 B ; hout (B,128) f32 = 524,288 B
    unsigned short* act = (unsigned short*)d_ws;
    float* hout = (float*)((char*)d_ws + (size_t)100 * 1024 * 64 * 2);

    conv_kernel<<<dim3(1024, 25), 256, 0, stream>>>(x, conv_w, conv_b, act);
    lstm_kernel<<<dim3(64), 512, 0, stream>>>(act, w_ih, w_hh, b_ih, b_hh, hout);
    inter_kernel<<<dim3(64), 256, 0, stream>>>(hout, W1, W2, W0, ln_g, ln_b, lin_w, lin_b, out);
}
